// DiseaseKnowledgeModule_86320252715156
// MI455X (gfx1250) — compile-verified, matrix operand order checked by reading, not run on hardware
//
#include <hip/hip_runtime.h>
#include <math.h>

typedef __bf16 bf16_t;
typedef __attribute__((ext_vector_type(16))) __bf16 v16bf;
typedef __attribute__((ext_vector_type(8)))  __bf16 v8bf;
typedef __attribute__((ext_vector_type(8)))  float  v8f;

#define B_    32
#define S_    1024
#define D_    1024
#define ND    14
#define MROWS (B_*S_)   // 32768

#define BM 128
#define BN 128
#define BK 32

// ---------------- kernel 0: zero the mlc_probs tail of d_out ----------------
__global__ void k_init_tail(float* __restrict__ tail) {
    tail[threadIdx.x] = 0.0f;
}

// ---- kernel 1: scores -> pairwise softmax P (workspace), atomic max -> tail ----
__global__ __launch_bounds__(256) void k_score(const float* __restrict__ z,
                                               const float* __restrict__ dk,
                                               float* __restrict__ P,
                                               float* __restrict__ tail) {
    __shared__ __align__(16) float sK[28][1024];   // 112 KB: all memory rows
    const int t = threadIdx.x;
    for (int idx = t; idx < 28 * 1024; idx += 256)
        sK[idx >> 10][idx & 1023] = dk[idx];
    __syncthreads();

    const int lane = t & 31;
    const int wave = t >> 5;
    for (int r = 0; r < 8; ++r) {                  // 8 rows per wave, 64 per block
        const int row = blockIdx.x * 64 + wave * 8 + r;
        const float4* z4 = (const float4*)(z + (size_t)row * D_);
        float acc[28];
        #pragma unroll
        for (int j = 0; j < 28; ++j) acc[j] = 0.0f;
        #pragma unroll
        for (int it = 0; it < 8; ++it) {           // 8 * 32 lanes * float4 = 1024
            const float4 zv = z4[it * 32 + lane];
            #pragma unroll
            for (int j = 0; j < 28; ++j) {
                const float4 kv = *(const float4*)&sK[j][(it * 32 + lane) * 4];
                acc[j] += zv.x * kv.x + zv.y * kv.y + zv.z * kv.z + zv.w * kv.w;
            }
        }
        #pragma unroll
        for (int j = 0; j < 28; ++j) {
            #pragma unroll
            for (int off = 16; off > 0; off >>= 1)
                acc[j] += __shfl_xor(acc[j], off, 32);
        }
        if (lane == 0) {
            const int b = row >> 10;
            #pragma unroll
            for (int i = 0; i < ND; ++i) {
                // softmax over 2 states == sigmoid of scaled score difference
                const float p = 1.0f / (1.0f + __expf((acc[2*i] - acc[2*i + 1]) * 0.03125f));
                P[row * ND + i] = p;
                // p > 0 always, so integer max on bit pattern == float max
                atomicMax((int*)(tail + b * ND + i), __float_as_int(p));
            }
        }
    }
}

// ---- kernel 2: G[i][d] = sum_k M_present[i][k] * W[d][1024+k]  (14 x 1024) ----
__global__ __launch_bounds__(256) void k_gmat(const float* __restrict__ dk,
                                              const float* __restrict__ gw,
                                              float* __restrict__ G) {
    const int wid  = (blockIdx.x * 256 + threadIdx.x) >> 5;  // 0..14335
    const int lane = threadIdx.x & 31;
    const int i = wid >> 10;        // disease
    const int d = wid & 1023;       // output column
    const float4* a4 = (const float4*)(dk + (size_t)(2 * i + 1) * D_);
    const float4* b4 = (const float4*)(gw + (size_t)d * (2 * D_) + D_);
    float acc = 0.0f;
    #pragma unroll
    for (int it = 0; it < 8; ++it) {
        const float4 a = a4[it * 32 + lane];
        const float4 b = b4[it * 32 + lane];
        acc += a.x * b.x + a.y * b.y + a.z * b.z + a.w * b.w;
    }
    #pragma unroll
    for (int off = 16; off > 0; off >>= 1) acc += __shfl_xor(acc, off, 32);
    if (lane == 0) G[i * D_ + d] = acc;
}

// ---- kernel 3: fused gate GEMM (bf16 WMMA) + sigmoid + residual output ----
__global__ __launch_bounds__(256) void k_gate(const float* __restrict__ z,
                                              const float* __restrict__ dk,
                                              const float* __restrict__ gw,
                                              const float* __restrict__ gb,
                                              const float* __restrict__ P,
                                              const float* __restrict__ G,
                                              float* __restrict__ out) {
    __shared__ __align__(16) bf16_t sA[BM][BK + 8];     // 10 KB
    __shared__ __align__(16) bf16_t sB[BN][BK + 8];     // 10 KB
    __shared__ __align__(16) float  sMp[ND][BN + 4];    // 7.4 KB
    __shared__ __align__(16) float  sGt[ND][BN + 4];    // 7.4 KB
    __shared__ __align__(16) float  sR [BM][BN + 4];    // 67.6 KB

    const int t    = threadIdx.x;
    const int lane = t & 31;
    const int wave = t >> 5;
    const int m0   = blockIdx.y * BM;
    const int n0   = blockIdx.x * BN;

    // stage this block's columns of M_present and G
    for (int idx = t; idx < ND * BN; idx += 256) {
        const int i = idx >> 7, c = idx & 127;
        sMp[i][c] = dk[(size_t)(2 * i + 1) * D_ + n0 + c];
        sGt[i][c] = G[i * D_ + n0 + c];
    }

    const int row = t >> 1;          // staging row (0..127), 2 threads per row
    const int c0  = (t & 1) * 16;    // staging K-column base
    float pr[ND];
    #pragma unroll
    for (int i = 0; i < ND; ++i) pr[i] = P[(size_t)(m0 + row) * ND + i];
    __syncthreads();

    // precompute R tile in LDS: sR[row][.] = sum_i P[row][i] * Mp[i][.]
    {
        const int cb0 = (t & 1) * 64;
        #pragma unroll
        for (int cc = 0; cc < 64; cc += 16) {
            float a[16];
            #pragma unroll
            for (int q = 0; q < 16; ++q) a[q] = 0.0f;
            #pragma unroll
            for (int i = 0; i < ND; ++i) {
                const float pv = pr[i];
                #pragma unroll
                for (int q = 0; q < 16; q += 4) {
                    const float4 mv = *(const float4*)&sMp[i][cb0 + cc + q];
                    a[q + 0] += pv * mv.x;  a[q + 1] += pv * mv.y;
                    a[q + 2] += pv * mv.z;  a[q + 3] += pv * mv.w;
                }
            }
            #pragma unroll
            for (int q = 0; q < 16; q += 4) {
                float4 st; st.x = a[q]; st.y = a[q+1]; st.z = a[q+2]; st.w = a[q+3];
                *(float4*)&sR[row][cb0 + cc + q] = st;
            }
        }
    }

    const int wm = (wave & 3) * 32;   // wave M offset inside block tile
    const int wn = (wave >> 2) * 64;  // wave N offset inside block tile

    v8f acc[2][4];
    const v8f zero8 = {0, 0, 0, 0, 0, 0, 0, 0};
    #pragma unroll
    for (int mt = 0; mt < 2; ++mt)
        #pragma unroll
        for (int nt = 0; nt < 4; ++nt) acc[mt][nt] = zero8;

    // register double-buffer for global -> LDS staging
    const float* baseA = z  + (size_t)(m0 + row) * D_       + c0;
    const float* baseB = gw + (size_t)(n0 + row) * (2 * D_) + c0;
    float4 bufA[4], bufB[4];
    auto prefetch = [&](int kc) {
        const float4* a = (const float4*)(baseA + kc);
        const float4* b = (const float4*)(baseB + kc);
        #pragma unroll
        for (int j = 0; j < 4; ++j) { bufA[j] = a[j]; bufB[j] = b[j]; }
    };
    prefetch(0);

    // 32 K-chunks of z/W1 + 1 zero-padded chunk carrying the rank-14 P*G term
    for (int kk = 0; kk < 33; ++kk) {
        __syncthreads();
        if (kk < 32) {
            #pragma unroll
            for (int j = 0; j < 4; ++j) {
                sA[row][c0+4*j+0] = (bf16_t)bufA[j].x; sA[row][c0+4*j+1] = (bf16_t)bufA[j].y;
                sA[row][c0+4*j+2] = (bf16_t)bufA[j].z; sA[row][c0+4*j+3] = (bf16_t)bufA[j].w;
                sB[row][c0+4*j+0] = (bf16_t)bufB[j].x; sB[row][c0+4*j+1] = (bf16_t)bufB[j].y;
                sB[row][c0+4*j+2] = (bf16_t)bufB[j].z; sB[row][c0+4*j+3] = (bf16_t)bufB[j].w;
            }
        } else {
            if (c0 == 0) {
                #pragma unroll
                for (int c = 0; c < ND; ++c) {
                    sA[row][c] = (bf16_t)pr[c];        // A ext. chunk = P
                    sB[row][c] = (bf16_t)sGt[c][row];  // B ext. chunk = G^T
                }
                sA[row][14] = (bf16_t)0.0f; sA[row][15] = (bf16_t)0.0f;
                sB[row][14] = (bf16_t)0.0f; sB[row][15] = (bf16_t)0.0f;
            } else {
                #pragma unroll
                for (int c = 16; c < 32; ++c) {
                    sA[row][c] = (bf16_t)0.0f;
                    sB[row][c] = (bf16_t)0.0f;
                }
            }
        }
        __syncthreads();

        // prefetch next K-chunk while this chunk's WMMAs execute
        if (kk + 1 < 32) prefetch((kk + 1) * BK);

        // ISA 16-bit A layout: lane L holds row (L&15), K chunks at (L>>4)*8, +16
        const int o = (lane >> 4) * 8;
        union Frag { v16bf v; v8bf h[2]; };
        Frag af[2], bfr[4];
        #pragma unroll
        for (int mt = 0; mt < 2; ++mt) {
            const int rr = wm + mt * 16 + (lane & 15);
            af[mt].h[0] = *(const v8bf*)&sA[rr][o];
            af[mt].h[1] = *(const v8bf*)&sA[rr][o + 16];
        }
        #pragma unroll
        for (int nt = 0; nt < 4; ++nt) {
            const int rr = wn + nt * 16 + (lane & 15);
            bfr[nt].h[0] = *(const v8bf*)&sB[rr][o];
            bfr[nt].h[1] = *(const v8bf*)&sB[rr][o + 16];
        }
        #pragma unroll
        for (int mt = 0; mt < 2; ++mt)
            #pragma unroll
            for (int nt = 0; nt < 4; ++nt)
                acc[mt][nt] = __builtin_amdgcn_wmma_f32_16x16x32_bf16(
                    false, af[mt].v, false, bfr[nt].v,
                    (short)0, acc[mt][nt], false, false);
    }

    // epilogue: out = z + sigmoid(acc + bias) * R
    #pragma unroll
    for (int mt = 0; mt < 2; ++mt) {
        #pragma unroll
        for (int nt = 0; nt < 4; ++nt) {
            const int n_loc = wn + nt * 16 + (lane & 15);
            const float bias = gb[n0 + n_loc];
            #pragma unroll
            for (int r = 0; r < 8; ++r) {
                // C/D layout: VGPR r -> M=r (lanes 0-15) or M=r+8 (lanes 16-31)
                const int m_loc = wm + mt * 16 + r + ((lane >> 4) << 3);
                const float logit = acc[mt][nt][r] + bias;
                const float g  = 1.0f / (1.0f + __expf(-logit));
                const float Rv = sR[m_loc][n_loc];
                const size_t gi = (size_t)(m0 + m_loc) * D_ + (size_t)(n0 + n_loc);
                out[gi] = z[gi] + g * Rv;
            }
        }
    }
}

extern "C" void kernel_launch(void* const* d_in, const int* in_sizes, int n_in,
                              void* d_out, int out_size, void* d_ws, size_t ws_size,
                              hipStream_t stream) {
    (void)in_sizes; (void)n_in; (void)out_size; (void)ws_size;
    const float* z  = (const float*)d_in[0];   // (32,1024,1024) f32
    const float* dk = (const float*)d_in[1];   // (14,2,1024)    f32
    const float* gw = (const float*)d_in[2];   // (1024,2048)    f32
    const float* gb = (const float*)d_in[3];   // (1024,)        f32

    float* out  = (float*)d_out;
    float* tail = out + (size_t)MROWS * D_;    // mlc_probs (32*14)

    float* P = (float*)d_ws;                   // 32768 x 14 f32 (~1.75 MB)
    float* G = P + (size_t)MROWS * ND;         // 14 x 1024 f32

    k_init_tail<<<1, B_ * ND, 0, stream>>>(tail);
    k_score<<<MROWS / 64, 256, 0, stream>>>(z, dk, P, tail);
    k_gmat<<<(ND * D_ * 32) / 256, 256, 0, stream>>>(dk, gw, G);

    dim3 grid(D_ / BN, MROWS / BM);
    k_gate<<<grid, 256, 0, stream>>>(z, dk, gw, gb, P, G, out);
}
